// InstanceDeformableConvBlock_30846455120523
// MI455X (gfx1250) — matrix operand order repaired and checked by reading, not compiled, not run on hardware
//
#include <hip/hip_runtime.h>
#include <hip/hip_bf16.h>

// ---------------------------------------------------------------------------
// Deformable-conv residual block on MI455X (gfx1250), wave32 + WMMA bf16.
// All four convolutions run as implicit GEMMs through
// v_wmma_f32_16x16x32_bf16 (fp32 activations downconverted to bf16 in LDS,
// f32 accumulation).  Double-buffered LDS tiles: one workgroup barrier per
// K-step, next tile's global loads / ds_stores co-issue with the WMMAs.
// The modulated bilinear gather of the deformable conv is fused into the
// B-tile fill of its GEMM.
// ---------------------------------------------------------------------------

typedef __attribute__((ext_vector_type(16))) __bf16 v16bf;
typedef __attribute__((ext_vector_type(8)))  __bf16 v8bf;
typedef __attribute__((ext_vector_type(8)))  float  v8f;

#define KCONV     2304          // 256*9
#define LDSA      40            // padded LDS row stride (bf16 elems), 80B: 16B-aligned
#define LDSB      40

__device__ __forceinline__ unsigned short f32_to_bf16(float f) {
    unsigned int u = __builtin_bit_cast(unsigned int, f);
    unsigned int r = u + 0x7FFFu + ((u >> 16) & 1u);   // round-to-nearest-even
    return (unsigned short)(r >> 16);
}

// A fragment (16x32 bf16): lane<16 -> M=lane, K {0..7,16..23}; lane>=16 -> K {8..15,24..31}
__device__ __forceinline__ v16bf load_frag_a(const unsigned short* As, int mBase, int lane) {
    int m  = mBase + (lane & 15);
    int ks = (lane >> 4) * 8;
    v8bf lo = *(const v8bf*)(As + m * LDSA + ks);
    v8bf hi = *(const v8bf*)(As + m * LDSA + ks + 16);
    return __builtin_shufflevector(lo, hi, 0,1,2,3,4,5,6,7,8,9,10,11,12,13,14,15);
}

// B fragment (32x16 bf16), B stored transposed in LDS as Bt[n][k]:
// lane<16 -> N=lane, K {0..7,16..23}; lane>=16 -> N=lane-16, K {8..15,24..31}
__device__ __forceinline__ v16bf load_frag_b(const unsigned short* Bs, int nBase, int lane) {
    int n  = nBase + (lane & 15);
    int ks = (lane >> 4) * 8;
    v8bf lo = *(const v8bf*)(Bs + n * LDSB + ks);
    v8bf hi = *(const v8bf*)(Bs + n * LDSB + ks + 16);
    return __builtin_shufflevector(lo, hi, 0,1,2,3,4,5,6,7,8,9,10,11,12,13,14,15);
}

// ---------------------------------------------------------------------------
// Weight prep
// ---------------------------------------------------------------------------
__global__ void cvt_f32_bf16(const float* __restrict__ src,
                             unsigned short* __restrict__ dst, int n) {
    int i = blockIdx.x * 256 + threadIdx.x;
    if (i < n) dst[i] = f32_to_bf16(src[i]);
}

// Stack offset conv (18 rows) + mod conv (9 rows) into one 128-row bf16 A.
__global__ void build_aom(const float* __restrict__ offw,
                          const float* __restrict__ modw,
                          unsigned short* __restrict__ dst) {
    int i = blockIdx.x * 256 + threadIdx.x;        // 128*2304 elements
    if (i >= 128 * KCONV) return;
    int row = i / KCONV, col = i - row * KCONV;
    float v = 0.f;
    if (row < 18)      v = offw[row * KCONV + col];
    else if (row < 27) v = modw[(row - 18) * KCONV + col];
    dst[i] = f32_to_bf16(v);
}

// ---------------------------------------------------------------------------
// Implicit-GEMM 3x3 conv (pad 1).  MODE 0: conv1 (+bias+BN1+ReLU -> out1)
//                                  MODE 1: offset/mod heads
// Block: 256 thr = 8 waves; tile 128(M) x 128(N); wave = 32x64 (2x4 WMMA accs)
// ---------------------------------------------------------------------------
template<int MODE>
__global__ __launch_bounds__(256) void conv3x3_wmma(
    const float* __restrict__ in,               // [4,256,64,64] fp32 NCHW
    const unsigned short* __restrict__ Aw,      // bf16 [Mrows][2304]
    const float* __restrict__ bias,             // conv1_b | off_b
    const float* __restrict__ bias2,            //  -      | mod_b
    const float* __restrict__ bng, const float* __restrict__ bnb,
    const float* __restrict__ bnm, const float* __restrict__ bnv,
    float* __restrict__ out0,                   // out1 | offset buf
    float* __restrict__ out1p)                  //  -   | mask buf
{
    __shared__ __align__(16) unsigned short As[2][128 * LDSA];
    __shared__ __align__(16) unsigned short Bs[2][128 * LDSB];

    const int tid   = threadIdx.x;
    const int lane  = tid & 31;
    const int wave  = tid >> 5;
    const int waveM = wave & 3;             // m base = 32*waveM
    const int waveN = wave >> 2;            // n base = 64*waveN
    const int nBase = blockIdx.x * 128;
    const int mBase = blockIdx.y * 128;

    // per-thread fill coordinates (one n column, 16 consecutive k)
    const int rA   = tid >> 1;
    const int cA   = (tid & 1) * 16;
    const int nnF  = tid >> 1;
    const int kbF  = (tid & 1) * 16;
    const int nF   = nBase + nnF;
    const int bF   = nF >> 12;
    const int pixF = nF & 4095;
    const int yF   = pixF >> 6;
    const int xF   = pixF & 63;

    auto fillA = [&](int k0, int buf) {
        const uint4* src = (const uint4*)(Aw + (size_t)(mBase + rA) * KCONV + k0 + cA);
        uint4* dst = (uint4*)&As[buf][rA * LDSA + cA];
        dst[0] = src[0];
        dst[1] = src[1];
        if (k0 + 32 < KCONV)   // gfx1250 global_prefetch_b8 for the next weight tile
            __builtin_prefetch(Aw + (size_t)(mBase + rA) * KCONV + k0 + 32 + cA, 0, 0);
    };
    auto fillB = [&](int k0, int buf) {
        #pragma unroll
        for (int j = 0; j < 16; ++j) {
            int kk = kbF + j;
            int k  = k0 + kk;
            int ci = k / 9;
            int t  = k - ci * 9;
            int ky = t / 3;
            int kx = t - ky * 3;
            int iy = yF + ky - 1;
            int ix = xF + kx - 1;
            float v = 0.f;
            if ((unsigned)iy < 64u && (unsigned)ix < 64u)
                v = in[(((bF << 8) + ci) << 12) + (iy << 6) + ix];
            Bs[buf][nnF * LDSB + kk] = f32_to_bf16(v);
        }
    };

    v8f acc[2][4];
    const v8f vzero = {0.f,0.f,0.f,0.f,0.f,0.f,0.f,0.f};
    #pragma unroll
    for (int i = 0; i < 2; ++i)
        #pragma unroll
        for (int j = 0; j < 4; ++j) acc[i][j] = vzero;

    fillA(0, 0);
    fillB(0, 0);
    int p = 0;
    for (int k0 = 0; k0 < KCONV; k0 += 32) {
        __syncthreads();
        v16bf afr[2], bfr[4];
        #pragma unroll
        for (int mi = 0; mi < 2; ++mi) afr[mi] = load_frag_a(As[p], waveM * 32 + mi * 16, lane);
        #pragma unroll
        for (int ni = 0; ni < 4; ++ni) bfr[ni] = load_frag_b(Bs[p], waveN * 64 + ni * 16, lane);
        #pragma unroll
        for (int mi = 0; mi < 2; ++mi)
            #pragma unroll
            for (int ni = 0; ni < 4; ++ni)
                acc[mi][ni] = __builtin_amdgcn_wmma_f32_16x16x32_bf16(
                    false, afr[mi], false, bfr[ni], (short)0, acc[mi][ni], false, false);
        int kn = k0 + 32;
        if (kn < KCONV) {      // fill the other buffer while WMMAs drain
            fillA(kn, p ^ 1);
            fillB(kn, p ^ 1);
        }
        p ^= 1;
    }

    // --- epilogue ---
    const int hlf = lane >> 4;
    const int nIn = lane & 15;
    #pragma unroll
    for (int mi = 0; mi < 2; ++mi)
        #pragma unroll
        for (int v = 0; v < 8; ++v) {
            int m = mBase + waveM * 32 + mi * 16 + hlf * 8 + v;
            if (MODE == 0) {
                float s  = bng[m] * rsqrtf(bnv[m] + 1e-5f);
                float sh = (bnb[m] - bnm[m] * s) + bias[m] * s;
                #pragma unroll
                for (int ni = 0; ni < 4; ++ni) {
                    int n   = nBase + waveN * 64 + ni * 16 + nIn;
                    int b   = n >> 12;
                    int pix = n & 4095;
                    float val = acc[mi][ni][v] * s + sh;
                    out0[(((b << 8) + m) << 12) + pix] = fmaxf(val, 0.f);
                }
            } else {
                if (m < 18) {
                    float bb = bias[m];
                    #pragma unroll
                    for (int ni = 0; ni < 4; ++ni) {
                        int n   = nBase + waveN * 64 + ni * 16 + nIn;
                        int b   = n >> 12;
                        int pix = n & 4095;
                        out0[((b * 18 + m) << 12) + pix] = acc[mi][ni][v] + bb;
                    }
                } else if (m < 27) {
                    float bb = bias2[m - 18];
                    #pragma unroll
                    for (int ni = 0; ni < 4; ++ni) {
                        int n   = nBase + waveN * 64 + ni * 16 + nIn;
                        int b   = n >> 12;
                        int pix = n & 4095;
                        float mm = acc[mi][ni][v] + bb;
                        out1p[((b * 9 + (m - 18)) << 12) + pix] = 2.f / (1.f + __expf(-mm));
                    }
                }
            }
        }
}

// ---------------------------------------------------------------------------
// Deformable conv implicit GEMM.  B tile is produced by modulated bilinear
// sampling of out1; the 9x128 tap table (clamped corners + mask-folded
// weights) is precomputed per block in LDS.  Epilogue applies BN2.
// ---------------------------------------------------------------------------
__global__ __launch_bounds__(256) void deform_wmma(
    const float* __restrict__ out1,             // [4,256,64,64]
    const unsigned short* __restrict__ A2,      // bf16 [256][2304]
    const float* __restrict__ offs,             // [4,18,64,64]
    const float* __restrict__ maskb,            // [4,9,64,64]
    const float* __restrict__ g2, const float* __restrict__ b2,
    const float* __restrict__ m2, const float* __restrict__ v2,
    float* __restrict__ out2bn)                 // [4,256,64,64]
{
    __shared__ __align__(16) unsigned short As[2][128 * LDSA];
    __shared__ __align__(16) unsigned short Bs[2][128 * LDSB];
    __shared__ float tw[9 * 128 * 4];           // w00,w01,w10,w11 (mask folded)
    __shared__ int   tc[9 * 128];               // packed clamped corners

    const int tid   = threadIdx.x;
    const int lane  = tid & 31;
    const int wave  = tid >> 5;
    const int waveM = wave & 3;
    const int waveN = wave >> 2;
    const int nBase = blockIdx.x * 128;
    const int mBase = blockIdx.y * 128;

    // ---- tap table: 9 taps x 128 columns ----
    for (int idx = tid; idx < 9 * 128; idx += 256) {
        int t  = idx >> 7;            // 0..8
        int nn = idx & 127;
        int n  = nBase + nn;
        int b  = n >> 12;
        int pix = n & 4095;
        int y  = pix >> 6;
        int xw = pix & 63;
        int ky = t / 3, kx = t - ky * 3;
        float dy = offs[((b * 18 + 2 * t) << 12) + pix];
        float dx = offs[((b * 18 + 2 * t + 1) << 12) + pix];
        float mv = maskb[((b * 9 + t) << 12) + pix];
        float py = (float)(y + ky - 1) + dy;
        float px = (float)(xw + kx - 1) + dx;
        float fy0 = floorf(py), fx0 = floorf(px);
        int y0 = (int)fy0, x0 = (int)fx0;
        int y1 = y0 + 1,  x1 = x0 + 1;
        float ly = py - fy0, lx = px - fx0;
        float w00 = (1.f - ly) * (1.f - lx) * mv;
        float w01 = (1.f - ly) * lx * mv;
        float w10 = ly * (1.f - lx) * mv;
        float w11 = ly * lx * mv;
        bool vy0 = (y0 >= 0) && (y0 <= 63);
        bool vy1 = (y1 >= 0) && (y1 <= 63);
        bool vx0 = (x0 >= 0) && (x0 <= 63);
        bool vx1 = (x1 >= 0) && (x1 <= 63);
        if (!(vy0 && vx0)) w00 = 0.f;
        if (!(vy0 && vx1)) w01 = 0.f;
        if (!(vy1 && vx0)) w10 = 0.f;
        if (!(vy1 && vx1)) w11 = 0.f;
        y0 = min(max(y0, 0), 63); x0 = min(max(x0, 0), 63);
        y1 = min(max(y1, 0), 63); x1 = min(max(x1, 0), 63);
        float* w = &tw[idx * 4];
        w[0] = w00; w[1] = w01; w[2] = w10; w[3] = w11;
        tc[idx] = y0 | (x0 << 8) | (y1 << 16) | (x1 << 24);
    }

    const int rA   = tid >> 1;
    const int cA   = (tid & 1) * 16;
    const int nnF  = tid >> 1;
    const int kbF  = (tid & 1) * 16;
    const int nF   = nBase + nnF;
    const int bF   = nF >> 12;

    auto fillA = [&](int k0, int buf) {
        const uint4* src = (const uint4*)(A2 + (size_t)(mBase + rA) * KCONV + k0 + cA);
        uint4* dst = (uint4*)&As[buf][rA * LDSA + cA];
        dst[0] = src[0];
        dst[1] = src[1];
        if (k0 + 32 < KCONV)
            __builtin_prefetch(A2 + (size_t)(mBase + rA) * KCONV + k0 + 32 + cA, 0, 0);
    };
    auto fillB = [&](int k0, int buf) {
        #pragma unroll
        for (int j = 0; j < 16; ++j) {
            int kk = kbF + j;
            int k  = k0 + kk;
            int ci = k / 9;
            int t  = k - ci * 9;
            int e  = t * 128 + nnF;
            int cc = tc[e];
            const float* w  = &tw[e * 4];
            int y0 =  cc        & 0xff;
            int x0 = (cc >> 8)  & 0xff;
            int y1 = (cc >> 16) & 0xff;
            int x1 = (cc >> 24) & 0xff;
            const float* pl = out1 + (((bF << 8) + ci) << 12);
            float v = w[0] * pl[(y0 << 6) + x0] + w[1] * pl[(y0 << 6) + x1]
                    + w[2] * pl[(y1 << 6) + x0] + w[3] * pl[(y1 << 6) + x1];
            Bs[buf][nnF * LDSB + kk] = f32_to_bf16(v);
        }
    };

    v8f acc[2][4];
    const v8f vzero = {0.f,0.f,0.f,0.f,0.f,0.f,0.f,0.f};
    #pragma unroll
    for (int i = 0; i < 2; ++i)
        #pragma unroll
        for (int j = 0; j < 4; ++j) acc[i][j] = vzero;

    __syncthreads();           // tap table ready
    fillA(0, 0);
    fillB(0, 0);
    int p = 0;
    for (int k0 = 0; k0 < KCONV; k0 += 32) {
        __syncthreads();
        v16bf afr[2], bfr[4];
        #pragma unroll
        for (int mi = 0; mi < 2; ++mi) afr[mi] = load_frag_a(As[p], waveM * 32 + mi * 16, lane);
        #pragma unroll
        for (int ni = 0; ni < 4; ++ni) bfr[ni] = load_frag_b(Bs[p], waveN * 64 + ni * 16, lane);
        #pragma unroll
        for (int mi = 0; mi < 2; ++mi)
            #pragma unroll
            for (int ni = 0; ni < 4; ++ni)
                acc[mi][ni] = __builtin_amdgcn_wmma_f32_16x16x32_bf16(
                    false, afr[mi], false, bfr[ni], (short)0, acc[mi][ni], false, false);
        int kn = k0 + 32;
        if (kn < KCONV) {
            fillA(kn, p ^ 1);
            fillB(kn, p ^ 1);
        }
        p ^= 1;
    }

    const int hlf = lane >> 4;
    const int nIn = lane & 15;
    #pragma unroll
    for (int mi = 0; mi < 2; ++mi)
        #pragma unroll
        for (int v = 0; v < 8; ++v) {
            int m = mBase + waveM * 32 + mi * 16 + hlf * 8 + v;
            float s  = g2[m] * rsqrtf(v2[m] + 1e-5f);
            float sh = b2[m] - m2[m] * s;
            #pragma unroll
            for (int ni = 0; ni < 4; ++ni) {
                int n   = nBase + waveN * 64 + ni * 16 + nIn;
                int b   = n >> 12;
                int pix = n & 4095;
                out2bn[(((b << 8) + m) << 12) + pix] = acc[mi][ni][v] * s + sh;
            }
        }
}

// ---------------------------------------------------------------------------
// 1x1 downsample GEMM + BN3 + residual add with out2bn + ReLU -> d_out
// ---------------------------------------------------------------------------
__global__ __launch_bounds__(256) void down_final_wmma(
    const float* __restrict__ x,                // [4,256,64,64]
    const unsigned short* __restrict__ A3,      // bf16 [256][256]
    const float* __restrict__ downb,
    const float* __restrict__ g3, const float* __restrict__ b3,
    const float* __restrict__ m3, const float* __restrict__ v3,
    const float* __restrict__ out2bn,
    float* __restrict__ outp)
{
    __shared__ __align__(16) unsigned short As[2][128 * LDSA];
    __shared__ __align__(16) unsigned short Bs[2][128 * LDSB];

    const int tid   = threadIdx.x;
    const int lane  = tid & 31;
    const int wave  = tid >> 5;
    const int waveM = wave & 3;
    const int waveN = wave >> 2;
    const int nBase = blockIdx.x * 128;
    const int mBase = blockIdx.y * 128;

    const int rA   = tid >> 1;
    const int cA   = (tid & 1) * 16;
    const int nnF  = tid >> 1;
    const int kbF  = (tid & 1) * 16;
    const int nF   = nBase + nnF;
    const int bF   = nF >> 12;
    const int pixF = nF & 4095;

    auto fillA = [&](int k0, int buf) {
        const uint4* src = (const uint4*)(A3 + (size_t)(mBase + rA) * 256 + k0 + cA);
        uint4* dst = (uint4*)&As[buf][rA * LDSA + cA];
        dst[0] = src[0];
        dst[1] = src[1];
        if (k0 + 32 < 256)
            __builtin_prefetch(A3 + (size_t)(mBase + rA) * 256 + k0 + 32 + cA, 0, 0);
    };
    auto fillB = [&](int k0, int buf) {
        #pragma unroll
        for (int j = 0; j < 16; ++j) {
            int kk = kbF + j;
            float v = x[(((bF << 8) + k0 + kk) << 12) + pixF];
            Bs[buf][nnF * LDSB + kk] = f32_to_bf16(v);
        }
    };

    v8f acc[2][4];
    const v8f vzero = {0.f,0.f,0.f,0.f,0.f,0.f,0.f,0.f};
    #pragma unroll
    for (int i = 0; i < 2; ++i)
        #pragma unroll
        for (int j = 0; j < 4; ++j) acc[i][j] = vzero;

    fillA(0, 0);
    fillB(0, 0);
    int p = 0;
    for (int k0 = 0; k0 < 256; k0 += 32) {
        __syncthreads();
        v16bf afr[2], bfr[4];
        #pragma unroll
        for (int mi = 0; mi < 2; ++mi) afr[mi] = load_frag_a(As[p], waveM * 32 + mi * 16, lane);
        #pragma unroll
        for (int ni = 0; ni < 4; ++ni) bfr[ni] = load_frag_b(Bs[p], waveN * 64 + ni * 16, lane);
        #pragma unroll
        for (int mi = 0; mi < 2; ++mi)
            #pragma unroll
            for (int ni = 0; ni < 4; ++ni)
                acc[mi][ni] = __builtin_amdgcn_wmma_f32_16x16x32_bf16(
                    false, afr[mi], false, bfr[ni], (short)0, acc[mi][ni], false, false);
        int kn = k0 + 32;
        if (kn < 256) {
            fillA(kn, p ^ 1);
            fillB(kn, p ^ 1);
        }
        p ^= 1;
    }

    const int hlf = lane >> 4;
    const int nIn = lane & 15;
    #pragma unroll
    for (int mi = 0; mi < 2; ++mi)
        #pragma unroll
        for (int v = 0; v < 8; ++v) {
            int m = mBase + waveM * 32 + mi * 16 + hlf * 8 + v;
            float s  = g3[m] * rsqrtf(v3[m] + 1e-5f);
            float sh = (b3[m] - m3[m] * s) + downb[m] * s;
            #pragma unroll
            for (int ni = 0; ni < 4; ++ni) {
                int n   = nBase + waveN * 64 + ni * 16 + nIn;
                int b   = n >> 12;
                int pix = n & 4095;
                size_t o = (size_t)(((b << 8) + m) << 12) + pix;
                float res = acc[mi][ni][v] * s + sh;
                outp[o] = fmaxf(out2bn[o] + res, 0.f);
            }
        }
}

// ---------------------------------------------------------------------------
extern "C" void kernel_launch(void* const* d_in, const int* in_sizes, int n_in,
                              void* d_out, int out_size, void* d_ws, size_t ws_size,
                              hipStream_t stream) {
    const float* x       = (const float*)d_in[0];
    const float* conv1_w = (const float*)d_in[1];
    const float* conv1_b = (const float*)d_in[2];
    const float* bn1_g   = (const float*)d_in[3];
    const float* bn1_b   = (const float*)d_in[4];
    const float* bn1_m   = (const float*)d_in[5];
    const float* bn1_v   = (const float*)d_in[6];
    const float* off_w   = (const float*)d_in[7];
    const float* off_b   = (const float*)d_in[8];
    const float* mod_w   = (const float*)d_in[9];
    const float* mod_b   = (const float*)d_in[10];
    const float* def_w   = (const float*)d_in[11];
    const float* bn2_g   = (const float*)d_in[12];
    const float* bn2_b   = (const float*)d_in[13];
    const float* bn2_m   = (const float*)d_in[14];
    const float* bn2_v   = (const float*)d_in[15];
    const float* down_w  = (const float*)d_in[16];
    const float* down_b  = (const float*)d_in[17];
    const float* bn3_g   = (const float*)d_in[18];
    const float* bn3_b   = (const float*)d_in[19];
    const float* bn3_m   = (const float*)d_in[20];
    const float* bn3_v   = (const float*)d_in[21];

    // workspace layout (requires ~38.5 MB)
    char* ws = (char*)d_ws;
    unsigned short* A1  = (unsigned short*)(ws + 0);          // 256x2304 bf16
    unsigned short* A2  = (unsigned short*)(ws + 1179648);    // 256x2304 bf16
    unsigned short* AOM = (unsigned short*)(ws + 2359296);    // 128x2304 bf16
    unsigned short* A3  = (unsigned short*)(ws + 2949120);    // 256x256  bf16
    float* out1   = (float*)(ws + 3080192);                   // 16.78 MB
    float* offs   = (float*)(ws + 19857408);                  // 1.18 MB
    float* maskb  = (float*)(ws + 21037056);                  // 0.59 MB
    float* out2bn = (float*)(ws + 21626880);                  // 16.78 MB

    cvt_f32_bf16<<<(256 * KCONV + 255) / 256, 256, 0, stream>>>(conv1_w, A1, 256 * KCONV);
    cvt_f32_bf16<<<(256 * KCONV + 255) / 256, 256, 0, stream>>>(def_w,   A2, 256 * KCONV);
    cvt_f32_bf16<<<(256 * 256 + 255) / 256,   256, 0, stream>>>(down_w,  A3, 256 * 256);
    build_aom   <<<(128 * KCONV + 255) / 256, 256, 0, stream>>>(off_w, mod_w, AOM);

    // conv1 + BN1 + ReLU
    conv3x3_wmma<0><<<dim3(128, 2), 256, 0, stream>>>(
        x, A1, conv1_b, nullptr, bn1_g, bn1_b, bn1_m, bn1_v, out1, nullptr);
    // offset + modulation heads
    conv3x3_wmma<1><<<dim3(128, 1), 256, 0, stream>>>(
        out1, AOM, off_b, mod_b, nullptr, nullptr, nullptr, nullptr, offs, maskb);
    // deformable conv + BN2
    deform_wmma<<<dim3(128, 2), 256, 0, stream>>>(
        out1, A2, offs, maskb, bn2_g, bn2_b, bn2_m, bn2_v, out2bn);
    // downsample + BN3 + residual + ReLU
    down_final_wmma<<<dim3(128, 2), 256, 0, stream>>>(
        x, A3, down_b, bn3_g, bn3_b, bn3_m, bn3_v, out2bn, (float*)d_out);
}
